// WorldModel_31645319037698
// MI455X (gfx1250) — matrix operand order repaired and checked by reading, not compiled, not run on hardware
//
#include <hip/hip_runtime.h>
#include <hip/hip_bf16.h>
#include <math.h>

// ---------------- model dims ----------------
static constexpr int B_  = 128;
static constexpr int T_  = 512;
static constexpr int SD  = 64;    // STATE_DIM
static constexpr int AD  = 16;    // ACTION_DIM
static constexpr int DETD = 512;  // DET
static constexpr int STD  = 64;   // STOCH
static constexpr int HIDD = 512;  // HID
static constexpr int KX   = 96;   // padded concat(stoch[64], act[16]) -> 96 (pad 80..95 = 0)

static constexpr int NBLK = 128;  // persistent grid (must be co-resident)
static constexpr int NTHR = 256;  // 8 wave32 per block

typedef __bf16 bf16_t;
typedef __attribute__((ext_vector_type(8)))  __bf16 v8bf;
typedef __attribute__((ext_vector_type(16))) __bf16 v16bf;
typedef __attribute__((ext_vector_type(8)))  float  v8f;

// ---------------- math helpers ----------------
__device__ inline float eluf(float x)      { return x > 0.f ? x : __expf(x) - 1.f; }
__device__ inline float sigmoidf_(float x) { return 1.f / (1.f + __expf(-x)); }
__device__ inline float softplusf_(float x){ return x > 20.f ? x : log1pf(__expf(x)); }

// ---------------- WMMA fragment loaders (wave32; this code's contiguous k order, same for A and B) ----
// A (16x32 bf16): lane l holds row M=l%16; khalf=l/16: elems 0..15 = K kBase+khalf*16..+15
__device__ inline v16bf load_a_frag(const bf16_t* rowBaseK, int khalf) {
  const v8bf* p0 = reinterpret_cast<const v8bf*>(rowBaseK + khalf * 16);
  const v8bf* p1 = reinterpret_cast<const v8bf*>(rowBaseK + khalf * 16 + 8);
  v8bf lo = *p0, hi = *p1;
  v16bf a;
#pragma unroll
  for (int i = 0; i < 8; ++i) { a[i] = lo[i]; a[8 + i] = hi[i]; }
  return a;
}
// B (32x16 bf16) built from row-major weight W[N][K] (so D = A * W^T):
// lane l covers col N=n; holds K = kBase + (l/16)*16 + 0..15 (16 contiguous bf16 of W row n)
__device__ inline v16bf load_b_frag(const bf16_t* W, int n, int Ktot, int kBase, int kgrp) {
  const bf16_t* base = W + (size_t)n * Ktot + kBase + kgrp * 16;
  const v8bf* p = reinterpret_cast<const v8bf*>(base);
  v8bf lo = p[0], hi = p[1];
  v16bf b;
#pragma unroll
  for (int i = 0; i < 8; ++i) { b[i] = lo[i]; b[8 + i] = hi[i]; }
  return b;
}

__device__ inline v8f zero8() {
  v8f a;
#pragma unroll
  for (int i = 0; i < 8; ++i) a[i] = 0.f;
  return a;
}

// one 16x16 tile, out = A(MxK) * W(NxK)^T, K = ksteps*32
__device__ inline v8f gemm_tile(const bf16_t* A, int aStride, int mt,
                                const bf16_t* W, int Ktot, int nt,
                                int ksteps, int lane) {
  const int mr = lane & 15, kg = lane >> 4;
  const bf16_t* aRow = A + (size_t)(mt * 16 + mr) * aStride;
  const int n = nt * 16 + mr;
  v8f acc = zero8();
  for (int kb = 0; kb < ksteps; ++kb) {
    const int kBase = kb * 32;
    v16bf a = load_a_frag(aRow + kBase, kg);
    v16bf b = load_b_frag(W, n, Ktot, kBase, kg);
    acc = __builtin_amdgcn_wmma_f32_16x16x32_bf16(false, a, false, b, (short)0, acc,
                                                  false, false);
  }
  return acc;
}

// ---------------- device-wide barrier for the persistent scan kernel ----------
__device__ inline void grid_barrier(unsigned* bar) {
  __threadfence();          // release: flush this thread's stores to L2 (device-coherent)
  __syncthreads();
  if (threadIdx.x == 0) {
    unsigned* cnt = bar;
    unsigned* gen = bar + 1;
    unsigned g = __hip_atomic_load(gen, __ATOMIC_RELAXED, __HIP_MEMORY_SCOPE_AGENT);
    unsigned arrived =
        __hip_atomic_fetch_add(cnt, 1u, __ATOMIC_ACQ_REL, __HIP_MEMORY_SCOPE_AGENT);
    if (arrived == (unsigned)gridDim.x - 1u) {
      __hip_atomic_store(cnt, 0u, __ATOMIC_RELAXED, __HIP_MEMORY_SCOPE_AGENT);
      __hip_atomic_fetch_add(gen, 1u, __ATOMIC_RELEASE, __HIP_MEMORY_SCOPE_AGENT);
    } else {
      while (__hip_atomic_load(gen, __ATOMIC_ACQUIRE, __HIP_MEMORY_SCOPE_AGENT) == g) {
        __builtin_amdgcn_s_sleep(1);
      }
    }
    __threadfence();        // acquire: invalidate this WGP's L0 before next phase reads
  }
  __syncthreads();
}

// ---------------- prep kernels ----------------
// fp32 -> bf16 with optional column padding (dcols >= scols, pad with 0)
__global__ void cvt_pad(const float* __restrict__ src, bf16_t* __restrict__ dst,
                        int rows, int scols, int dcols) {
  size_t total = (size_t)rows * dcols;
  for (size_t i = (size_t)blockIdx.x * blockDim.x + threadIdx.x; i < total;
       i += (size_t)gridDim.x * blockDim.x) {
    int r = (int)(i / dcols), c = (int)(i % dcols);
    dst[i] = (c < scols) ? (bf16_t)src[(size_t)r * scols + c] : (bf16_t)0.f;
  }
}

// obs embedding: emb = elu(obs_bf16 @ enc_w^T + enc_b), M=B*T=65536, N=512, K=64
__global__ void __launch_bounds__(256) emb_gemm(const bf16_t* __restrict__ obsb,
                                                const bf16_t* __restrict__ w_enc,
                                                const float* __restrict__ enc_b,
                                                bf16_t* __restrict__ emb) {
  const int lane = threadIdx.x & 31;
  const int wid = (blockIdx.x * blockDim.x + threadIdx.x) >> 5;
  const int totalTiles = (B_ * T_ / 16) * (HIDD / 16);   // 4096 * 32
  if (wid >= totalTiles) return;
  const int mt = wid >> 5, nt = wid & 31;
  v8f acc = gemm_tile(obsb, SD, mt, w_enc, SD, nt, SD / 32, lane);
  const int col = nt * 16 + (lane & 15);
  const float bias = enc_b[col];
  const int rb = mt * 16 + (lane >> 4) * 8;
#pragma unroll
  for (int r = 0; r < 8; ++r)
    emb[(size_t)(rb + r) * HIDD + col] = (bf16_t)eluf(acc[r] + bias);
}

// init: xin = [0(stoch) | act[:,0,:] | 0 pad], det = 0, barrier = 0
__global__ void init_state(bf16_t* __restrict__ xin, bf16_t* __restrict__ detb,
                           float* __restrict__ det32,
                           const float* __restrict__ actions, unsigned* __restrict__ bar) {
  size_t i = (size_t)blockIdx.x * blockDim.x + threadIdx.x;
  if (i < (size_t)B_ * KX) {
    int b = (int)(i / KX), c = (int)(i % KX);
    float v = 0.f;
    if (c >= STD && c < STD + AD) v = actions[(size_t)b * T_ * AD + (c - STD)]; // t = 0
    xin[i] = (bf16_t)v;
  }
  if (i < (size_t)B_ * DETD) { det32[i] = 0.f; detb[i] = (bf16_t)0.f; }
  if (i < 2) bar[i] = 0u;
}

// ---------------- persistent RSSM scan ----------------
struct ScanParams {
  const float* actions; const float* eps;
  const float* gru_bi; const float* gru_bh;
  const float* pri_enc_b; const float* pri_d1_b; const float* pri_d2_b;
  const float* pos_enc_b; const float* pos_dec_b;
  const bf16_t* w_prienc; const bf16_t* w_gwi; const bf16_t* w_gwh;
  const bf16_t* w_pd1; const bf16_t* w_pd2; const bf16_t* w_penc; const bf16_t* w_pdec;
  const bf16_t* emb;
  bf16_t* xin; bf16_t* xb; bf16_t* detb; bf16_t* px1; bf16_t* qx1;
  float* det32; float* gi; float* gh;
  float* o_stoch; float* o_det; float* o_qsh; float* o_qsc; float* o_psh; float* o_psc;
  unsigned* bar;
};

__global__ void __launch_bounds__(NTHR) rssm_scan(ScanParams p) {
  const int lane = threadIdx.x & 31;
  const int wid = (blockIdx.x * NTHR + threadIdx.x) >> 5;
  const int nwaves = (NBLK * NTHR) >> 5;           // 1024 wave32s
  const int tid = blockIdx.x * NTHR + threadIdx.x;
  const int nthreads = NBLK * NTHR;

  for (int t = 0; t < T_; ++t) {
    // ---- Phase A: x = elu(pri_enc([stoch, act])) ; [128x512], K=96 (padded)
    for (int tt = wid; tt < (B_ / 16) * (DETD / 16); tt += nwaves) {
      const int mt = tt >> 5, nt = tt & 31;
      v8f acc = gemm_tile(p.xin, KX, mt, p.w_prienc, KX, nt, KX / 32, lane);
      const int col = nt * 16 + (lane & 15);
      const float bias = p.pri_enc_b[col];
      const int rb = mt * 16 + (lane >> 4) * 8;
#pragma unroll
      for (int r = 0; r < 8; ++r)
        p.xb[(size_t)(rb + r) * DETD + col] = (bf16_t)eluf(acc[r] + bias);
    }
    grid_barrier(p.bar);

    // ---- Phase B1: gi = x @ Wi^T, gh = det @ Wh^T ; 2 x [128x1536] = 1536 tiles
    for (int tt = wid; tt < 1536; tt += nwaves) {
      const bool isGi = tt < 768;
      const int q = isGi ? tt : tt - 768;
      const int mt = q / 96, nt = q % 96;
      const bf16_t* A = isGi ? p.xb : p.detb;
      const bf16_t* W = isGi ? p.w_gwi : p.w_gwh;
      v8f acc = gemm_tile(A, DETD, mt, W, DETD, nt, DETD / 32, lane);
      float* dst = isGi ? p.gi : p.gh;
      const int col = nt * 16 + (lane & 15);
      const int rb = mt * 16 + (lane >> 4) * 8;
#pragma unroll
      for (int r = 0; r < 8; ++r) dst[(size_t)(rb + r) * 1536 + col] = acc[r];
    }
    grid_barrier(p.bar);

    // ---- Phase B2: GRU combine (elementwise, 128x512)
    for (int i = tid; i < B_ * DETD; i += nthreads) {
      const int b = i >> 9, c = i & 511;
      const float* gi = p.gi + (size_t)b * 1536;
      const float* gh = p.gh + (size_t)b * 1536;
      const float ir = gi[c] + p.gru_bi[c];
      const float hr = gh[c] + p.gru_bh[c];
      const float iz = gi[c + 512] + p.gru_bi[c + 512];
      const float hz = gh[c + 512] + p.gru_bh[c + 512];
      const float ia = gi[c + 1024] + p.gru_bi[c + 1024];
      const float ha = gh[c + 1024] + p.gru_bh[c + 1024];
      const float r = sigmoidf_(ir + hr);
      const float z = sigmoidf_(iz + hz);
      const float n = tanhf(ia + r * ha);
      const float d = (1.f - z) * n + z * p.det32[i];
      p.det32[i] = d;
      p.detb[i] = (bf16_t)d;
      p.o_det[((size_t)b * T_ + t) * DETD + c] = d;
    }
    grid_barrier(p.bar);

    // ---- Phase C: px1 = elu(det@W1^T+b), qx1 = elu([det,emb]@Wenc^T+b) ; 512 tiles
    for (int tt = wid; tt < 512; tt += nwaves) {
      if (tt < 256) {
        const int mt = tt >> 5, nt = tt & 31;
        v8f acc = gemm_tile(p.detb, DETD, mt, p.w_pd1, DETD, nt, DETD / 32, lane);
        const int col = nt * 16 + (lane & 15);
        const float bias = p.pri_d1_b[col];
        const int rb = mt * 16 + (lane >> 4) * 8;
#pragma unroll
        for (int r = 0; r < 8; ++r)
          p.px1[(size_t)(rb + r) * HIDD + col] = (bf16_t)eluf(acc[r] + bias);
      } else {
        const int q = tt - 256;
        const int mt = q >> 5, nt = q & 31;
        const int mr = lane & 15, kg = lane >> 4;
        const int m = mt * 16 + mr;          // batch row
        const int n = nt * 16 + mr;
        v8f acc = zero8();
        for (int kb = 0; kb < (DETD + HIDD) / 32; ++kb) {
          const int kBase = kb * 32;
          const bf16_t* aRow =
              (kBase < DETD) ? p.detb + (size_t)m * DETD + kBase
                             : p.emb + ((size_t)m * T_ + t) * HIDD + (kBase - DETD);
          v16bf a = load_a_frag(aRow, kg);
          v16bf b = load_b_frag(p.w_penc, n, DETD + HIDD, kBase, kg);
          acc = __builtin_amdgcn_wmma_f32_16x16x32_bf16(false, a, false, b, (short)0, acc,
                                                        false, false);
        }
        const int col = nt * 16 + (lane & 15);
        const float bias = p.pos_enc_b[col];
        const int rb = mt * 16 + kg * 8;
#pragma unroll
        for (int r = 0; r < 8; ++r)
          p.qx1[(size_t)(rb + r) * HIDD + col] = (bf16_t)eluf(acc[r] + bias);
      }
    }
    grid_barrier(p.bar);

    // ---- Phase D: heads 512->128 as (shift, raw) column pairs; fused epilogue
    if (wid < 64) {
      const bool prior = wid < 32;
      const int q = prior ? wid : wid - 32;
      const int mt = q >> 2, np = q & 3;
      const bf16_t* A = prior ? p.px1 : p.qx1;
      const bf16_t* W = prior ? p.w_pd2 : p.w_pdec;
      const float* bias = prior ? p.pri_d2_b : p.pos_dec_b;
      const int mr = lane & 15, kg = lane >> 4;
      const bf16_t* aRow = A + (size_t)(mt * 16 + mr) * HIDD;
      const int n0 = np * 16 + mr, n1 = STD + np * 16 + mr;
      v8f acc0 = zero8(), acc1 = zero8();
      for (int kb = 0; kb < HIDD / 32; ++kb) {
        const int kBase = kb * 32;
        v16bf a = load_a_frag(aRow + kBase, kg);
        v16bf b0 = load_b_frag(W, n0, HIDD, kBase, kg);
        v16bf b1 = load_b_frag(W, n1, HIDD, kBase, kg);
        acc0 = __builtin_amdgcn_wmma_f32_16x16x32_bf16(false, a, false, b0, (short)0, acc0,
                                                       false, false);
        acc1 = __builtin_amdgcn_wmma_f32_16x16x32_bf16(false, a, false, b1, (short)0, acc1,
                                                       false, false);
      }
      const int col = np * 16 + (lane & 15);          // 0..63
      const float bsh = bias[col], braw = bias[col + STD];
      const int rb = mt * 16 + kg * 8;
#pragma unroll
      for (int r = 0; r < 8; ++r) {
        const int b = rb + r;
        const size_t o = ((size_t)b * T_ + t) * STD + col;
        const float shift = acc0[r] + bsh;
        const float scale = softplusf_(acc1[r] + braw) + 0.1f;
        if (prior) {
          p.o_psh[o] = shift; p.o_psc[o] = scale;
        } else {
          p.o_qsh[o] = shift; p.o_qsc[o] = scale;
          const float s = shift + scale * p.eps[o];
          p.o_stoch[o] = s;
          p.xin[(size_t)b * KX + col] = (bf16_t)s;    // next step's stoch input
        }
      }
    } else if (wid < 128 && (t + 1) < T_) {
      // stage next step's action into xin cols 64..79 (128*16 = 2048 elems)
      const int idx = (wid - 64) * 32 + lane;
      const int b = idx >> 4, a2 = idx & 15;
      p.xin[(size_t)b * KX + STD + a2] =
          (bf16_t)p.actions[((size_t)b * T_ + (t + 1)) * AD + a2];
    }
    grid_barrier(p.bar);
  }
}

// ---------------- decoder: outs = [stoch, det] @ dec_w^T + dec_b ----------------
static constexpr int DEC_ROWS = 8;
__global__ void __launch_bounds__(256) dec_out(const float* __restrict__ stochs,
                                               const float* __restrict__ dets,
                                               const float* __restrict__ dec_w,
                                               const float* __restrict__ dec_b,
                                               float* __restrict__ outs) {
  __shared__ float lf[DEC_ROWS][STD + DETD];   // 8 x 576
  const size_t row0 = (size_t)blockIdx.x * DEC_ROWS;
  for (int i = threadIdx.x; i < DEC_ROWS * (STD + DETD); i += blockDim.x) {
    const int rr = i / (STD + DETD), k = i % (STD + DETD);
    const size_t row = row0 + rr;
    lf[rr][k] = (k < STD) ? stochs[row * STD + k] : dets[row * DETD + (k - STD)];
  }
  __syncthreads();
  for (int i = threadIdx.x; i < DEC_ROWS * (SD + 2); i += blockDim.x) {
    const int rr = i / (SD + 2), o = i % (SD + 2);
    const float* w = dec_w + (size_t)o * (STD + DETD);
    const float* f = lf[rr];
    float s = dec_b[o];
#pragma unroll 4
    for (int k = 0; k < STD + DETD; ++k) s += f[k] * w[k];
    outs[(row0 + rr) * (SD + 2) + o] = s;
  }
}

// ---------------- host-side orchestration ----------------
extern "C" void kernel_launch(void* const* d_in, const int* in_sizes, int n_in,
                              void* d_out, int out_size, void* d_ws, size_t ws_size,
                              hipStream_t stream) {
  (void)in_sizes; (void)n_in; (void)out_size; (void)ws_size;
  const float* observation = (const float*)d_in[0];
  const float* actions     = (const float*)d_in[1];
  const float* eps         = (const float*)d_in[2];
  const float* enc_w       = (const float*)d_in[3];
  const float* enc_b       = (const float*)d_in[4];
  const float* dec_w       = (const float*)d_in[5];
  const float* dec_b       = (const float*)d_in[6];
  const float* pri_enc_w   = (const float*)d_in[7];
  const float* pri_enc_b   = (const float*)d_in[8];
  const float* gru_wi      = (const float*)d_in[9];
  const float* gru_bi      = (const float*)d_in[10];
  const float* gru_wh      = (const float*)d_in[11];
  const float* gru_bh      = (const float*)d_in[12];
  const float* pri_d1_w    = (const float*)d_in[13];
  const float* pri_d1_b    = (const float*)d_in[14];
  const float* pri_d2_w    = (const float*)d_in[15];
  const float* pri_d2_b    = (const float*)d_in[16];
  const float* pos_enc_w   = (const float*)d_in[17];
  const float* pos_enc_b   = (const float*)d_in[18];
  const float* pos_dec_w   = (const float*)d_in[19];
  const float* pos_dec_b   = (const float*)d_in[20];

  // ---- d_out layout: (stochs, dets, outs, q_sh, q_sc, p_sh, p_sc) flat ----
  float* out = (float*)d_out;
  const size_t nBTS = (size_t)B_ * T_ * STD;   // 4,194,304
  float* o_stoch = out;
  float* o_det   = out + nBTS;
  float* o_outs  = o_det + (size_t)B_ * T_ * DETD;
  float* o_qsh   = o_outs + (size_t)B_ * T_ * (SD + 2);
  float* o_qsc   = o_qsh + nBTS;
  float* o_psh   = o_qsc + nBTS;
  float* o_psc   = o_psh + nBTS;

  // ---- workspace carve (L2-resident working set, ~83 MB) ----
  char* ws = (char*)d_ws;
  size_t off = 0;
  auto take = [&](size_t elems, size_t esz) -> void* {
    void* q = ws + off;
    off += (elems * esz + 255) & ~(size_t)255;
    return q;
  };
  bf16_t* w_prienc = (bf16_t*)take((size_t)DETD * KX, 2);
  bf16_t* w_gwi    = (bf16_t*)take((size_t)3 * DETD * DETD, 2);
  bf16_t* w_gwh    = (bf16_t*)take((size_t)3 * DETD * DETD, 2);
  bf16_t* w_pd1    = (bf16_t*)take((size_t)HIDD * DETD, 2);
  bf16_t* w_pd2    = (bf16_t*)take((size_t)2 * STD * HIDD, 2);
  bf16_t* w_penc   = (bf16_t*)take((size_t)HIDD * (DETD + HIDD), 2);
  bf16_t* w_pdec   = (bf16_t*)take((size_t)2 * STD * HIDD, 2);
  bf16_t* w_enc    = (bf16_t*)take((size_t)HIDD * SD, 2);
  bf16_t* obsb     = (bf16_t*)take((size_t)B_ * T_ * SD, 2);
  bf16_t* emb      = (bf16_t*)take((size_t)B_ * T_ * HIDD, 2);
  bf16_t* xin      = (bf16_t*)take((size_t)B_ * KX, 2);
  bf16_t* xb       = (bf16_t*)take((size_t)B_ * DETD, 2);
  bf16_t* detb     = (bf16_t*)take((size_t)B_ * DETD, 2);
  bf16_t* px1      = (bf16_t*)take((size_t)B_ * HIDD, 2);
  bf16_t* qx1      = (bf16_t*)take((size_t)B_ * HIDD, 2);
  float*  det32    = (float*)take((size_t)B_ * DETD, 4);
  float*  gi       = (float*)take((size_t)B_ * 3 * DETD, 4);
  float*  gh       = (float*)take((size_t)B_ * 3 * DETD, 4);
  unsigned* bar    = (unsigned*)take(2, 4);

  auto cvt = [&](const float* src, bf16_t* dst, int rows, int scols, int dcols) {
    int total = rows * dcols;
    int blocks = (total + 255) / 256;
    if (blocks > 2048) blocks = 2048;
    cvt_pad<<<blocks, 256, 0, stream>>>(src, dst, rows, scols, dcols);
  };
  // weights -> bf16 (pri_enc padded 80 -> 96 with zeros)
  cvt(pri_enc_w, w_prienc, DETD, STD + AD, KX);
  cvt(gru_wi,    w_gwi,    3 * DETD, DETD, DETD);
  cvt(gru_wh,    w_gwh,    3 * DETD, DETD, DETD);
  cvt(pri_d1_w,  w_pd1,    HIDD, DETD, DETD);
  cvt(pri_d2_w,  w_pd2,    2 * STD, HIDD, HIDD);
  cvt(pos_enc_w, w_penc,   HIDD, DETD + HIDD, DETD + HIDD);
  cvt(pos_dec_w, w_pdec,   2 * STD, HIDD, HIDD);
  cvt(enc_w,     w_enc,    HIDD, SD, SD);
  cvt(observation, obsb,   B_ * T_, SD, SD);

  // embedding GEMM (WMMA): 131072 tiles, 8 waves/block
  emb_gemm<<<(B_ * T_ / 16) * (HIDD / 16) / 8, 256, 0, stream>>>(obsb, w_enc, enc_b, emb);

  // init recurrent state + barrier
  init_state<<<256, 256, 0, stream>>>(xin, detb, det32, actions, bar);

  // persistent scan over T=512 steps
  ScanParams p;
  p.actions = actions; p.eps = eps;
  p.gru_bi = gru_bi; p.gru_bh = gru_bh;
  p.pri_enc_b = pri_enc_b; p.pri_d1_b = pri_d1_b; p.pri_d2_b = pri_d2_b;
  p.pos_enc_b = pos_enc_b; p.pos_dec_b = pos_dec_b;
  p.w_prienc = w_prienc; p.w_gwi = w_gwi; p.w_gwh = w_gwh;
  p.w_pd1 = w_pd1; p.w_pd2 = w_pd2; p.w_penc = w_penc; p.w_pdec = w_pdec;
  p.emb = emb;
  p.xin = xin; p.xb = xb; p.detb = detb; p.px1 = px1; p.qx1 = qx1;
  p.det32 = det32; p.gi = gi; p.gh = gh;
  p.o_stoch = o_stoch; p.o_det = o_det;
  p.o_qsh = o_qsh; p.o_qsc = o_qsc; p.o_psh = o_psh; p.o_psc = o_psc;
  p.bar = bar;
  rssm_scan<<<NBLK, NTHR, 0, stream>>>(p);

  // final decode
  dec_out<<<B_ * T_ / DEC_ROWS, 256, 0, stream>>>(o_stoch, o_det, dec_w, dec_b, o_outs);
}
